// BASE_RelPatchAttention2D_THW_67697274519585
// MI455X (gfx1250) — matrix operand order repaired and checked by reading, not compiled, not run on hardware
//
#include <hip/hip_runtime.h>
#include <hip/hip_bf16.h>

typedef __attribute__((ext_vector_type(16))) _Float16 v16h;
typedef __attribute__((ext_vector_type(8)))  _Float16 v8h;
typedef __attribute__((ext_vector_type(8)))  float    v8f;
typedef __attribute__((ext_vector_type(4)))  int      v4i;
typedef __attribute__((address_space(1))) v4i gv4i;   // global int4
typedef __attribute__((address_space(3))) v4i sv4i;   // LDS int4

#define NB 4            // batches
#define NP 4096         // patches per batch (t*16*16)
#define KD 1024         // inner dim per patch (c*h*w)
#define SMOOTH 1e-05f

#define BM 128
#define BN 256
#define BK 32
#define LDK 40          // padded LDS row stride (halves); 80B rows keep 16B alignment
#define NKSTEP (KD / BK)

// ---------------------------------------------------------------------------
// CDNA5 async global->LDS staging (ASYNCcnt path), with sync fallback.
// ---------------------------------------------------------------------------
__device__ __forceinline__ void async_copy16(const _Float16* g, _Float16* l) {
#if __has_builtin(__builtin_amdgcn_global_load_async_to_lds_b128)
    __builtin_amdgcn_global_load_async_to_lds_b128(
        (gv4i*)(void*)g, (sv4i*)(void*)l, 0, 0);
#else
    *(uint4*)l = *(const uint4*)g;
#endif
}

__device__ __forceinline__ void wait_async() {
#if __has_builtin(__builtin_amdgcn_global_load_async_to_lds_b128)
  #if __has_builtin(__builtin_amdgcn_s_wait_asynccnt)
    __builtin_amdgcn_s_wait_asynccnt(0);
  #else
    asm volatile("s_wait_asynccnt 0" ::: "memory");
  #endif
#endif
}

// ---------------------------------------------------------------------------
// Stage 1: patchify + f16 pack + squared-norm per patch.
// One 256-thread block per (batch, patch); 4 elements per thread.
// ---------------------------------------------------------------------------
__global__ __launch_bounds__(256) void pack_kernel(const float* __restrict__ x,
                                                   _Float16* __restrict__ Xp,
                                                   float* __restrict__ norms) {
    const int p     = blockIdx.x;            // b*4096 + patch
    const int b     = p >> 12;
    const int patch = p & 4095;
    const int t  = patch >> 8;
    const int ph = (patch >> 4) & 15;
    const int pw = patch & 15;
    const float* src = x + (size_t)(b * 16 + t) * 262144 + ph * 1024 + pw * 8;

    float ss = 0.f;
#pragma unroll
    for (int i = 0; i < 4; ++i) {
        const int kidx = threadIdx.x + (i << 8);
        const int c = kidx >> 6, n = (kidx >> 3) & 7, o = kidx & 7;
        const float vv = src[c * 16384 + n * 128 + o];
        Xp[(size_t)p * KD + kidx] = (_Float16)vv;
        ss += vv * vv;
    }
    __shared__ float red[256];
    red[threadIdx.x] = ss;
    __syncthreads();
    for (int s = 128; s > 0; s >>= 1) {
        if ((int)threadIdx.x < s) red[threadIdx.x] += red[threadIdx.x + s];
        __syncthreads();
    }
    if (threadIdx.x == 0) norms[p] = red[0];
}

__global__ void zero_kernel(float* __restrict__ p, int n) {
    const int i = blockIdx.x * blockDim.x + threadIdx.x;
    if (i < n) p[i] = 0.f;
}

// ---------------------------------------------------------------------------
// Stage 2: per-batch 4096x4096x1024 Gram GEMM (f16 WMMA, f32 accum),
// fused Jaccard similarity + column-sum reduction into tqk[b][key_patch].
// Workgroup: 256 threads = 8 wave32, tile 128x256, BK=32, double-buffered LDS
// filled via async global->LDS. Wave grid 2(M) x 4(N); each wave owns a
// 64x64 region = 4x4 WMMA 16x16 subtiles -> 16 v_wmma per k-step per wave.
// ---------------------------------------------------------------------------
__global__ __launch_bounds__(256, 1) void gemm_sim_kernel(const _Float16* __restrict__ A,
                                                          const _Float16* __restrict__ B,
                                                          const float* __restrict__ qq,
                                                          const float* __restrict__ kk,
                                                          float* __restrict__ tqk) {
    __shared__ _Float16 As[2][BM * LDK];
    __shared__ _Float16 Bs[2][BN * LDK];
    __shared__ float colsum[BN];

    const int b     = blockIdx.z;
    const int tileM = blockIdx.y * BM;
    const int tileN = blockIdx.x * BN;
    const _Float16* Ab = A + (size_t)b * NP * KD;
    const _Float16* Bb = B + (size_t)b * NP * KD;

    const int tid  = threadIdx.x;
    const int wid  = tid >> 5;
    const int lane = tid & 31;
    const int wm   = wid & 1;        // wave row  -> rows wm*64
    const int wn   = wid >> 1;       // wave col  -> cols wn*64
    const int lrow = lane & 15;
    const int half = lane >> 4;      // 0: low K group, 1: high K group

    // async global -> LDS tile fill: A 128x32 halves, B 256x32 halves
    auto issue_tile = [&](int buf, int ks) {
        const int r  = tid >> 2;               // 0..63
        const int ch = tid & 3;                // 16B chunk within a 64B row
        const size_t kof = (size_t)ks * BK + ch * 8;
#pragma unroll
        for (int i = 0; i < 2; ++i) {
            const int row = r + i * 64;
            async_copy16(Ab + (size_t)(tileM + row) * KD + kof,
                         &As[buf][row * LDK + ch * 8]);
        }
#pragma unroll
        for (int i = 0; i < 4; ++i) {
            const int row = r + i * 64;
            async_copy16(Bb + (size_t)(tileN + row) * KD + kof,
                         &Bs[buf][row * LDK + ch * 8]);
        }
    };

    v8f acc[4][4];
#pragma unroll
    for (int mi = 0; mi < 4; ++mi)
#pragma unroll
        for (int ni = 0; ni < 4; ++ni) acc[mi][ni] = (v8f)(0.f);

    colsum[tid] = 0.f;

    issue_tile(0, 0);
    wait_async();
    __syncthreads();

    for (int ks = 0; ks < NKSTEP; ++ks) {
        const int cur = ks & 1;
        if (ks + 1 < NKSTEP) issue_tile(cur ^ 1, ks + 1);

        // A fragments: lane<16 holds K{0..7,16..23}; lane>=16 holds K{8..15,24..31}
        v16h afrag[4];
#pragma unroll
        for (int mi = 0; mi < 4; ++mi) {
            const _Float16* p = &As[cur][(wm * 64 + mi * 16 + lrow) * LDK + half * 8];
            const v8h lo = *(const v8h*)p;
            const v8h hi = *(const v8h*)(p + 16);
            afrag[mi] = __builtin_shufflevector(lo, hi, 0, 1, 2, 3, 4, 5, 6, 7,
                                                8, 9, 10, 11, 12, 13, 14, 15);
        }
        // B fragments: lane<16 holds K{0..7,16..23}; lane>=16 holds K{8..15,24..31} (col = lane&15)
        v16h bfrag[4];
#pragma unroll
        for (int ni = 0; ni < 4; ++ni) {
            const _Float16* p = &Bs[cur][(wn * 64 + ni * 16 + lrow) * LDK + half * 8];
            const v8h lo = *(const v8h*)p;
            const v8h hi = *(const v8h*)(p + 16);
            bfrag[ni] = __builtin_shufflevector(lo, hi, 0, 1, 2, 3, 4, 5, 6, 7,
                                                8, 9, 10, 11, 12, 13, 14, 15);
        }
#pragma unroll
        for (int mi = 0; mi < 4; ++mi)
#pragma unroll
            for (int ni = 0; ni < 4; ++ni)
                acc[mi][ni] = __builtin_amdgcn_wmma_f32_16x16x32_f16(
                    false, afrag[mi], false, bfrag[ni], (short)0, acc[mi][ni],
                    false, false);

        wait_async();     // this wave's prefetch into buf^1 landed in LDS
        __syncthreads();  // whole workgroup ready to flip buffers
    }

    // Epilogue: sim = (qk+s)/(|q|^2+|k|^2-qk+s); reduce over rows -> colsum
    const int qbase = b * NP + tileM + wm * 64;
    const int hb    = half * 8;      // C layout: lanes>=16 hold M=8+j
    float qrow[32];
#pragma unroll
    for (int mi = 0; mi < 4; ++mi)
#pragma unroll
        for (int j = 0; j < 8; ++j)
            qrow[mi * 8 + j] = qq[qbase + mi * 16 + hb + j];

#pragma unroll
    for (int ni = 0; ni < 4; ++ni) {
        const int colLocal = wn * 64 + ni * 16 + lrow;
        const float kkv = kk[b * NP + tileN + colLocal];
        float csum = 0.f;
#pragma unroll
        for (int mi = 0; mi < 4; ++mi)
#pragma unroll
            for (int j = 0; j < 8; ++j) {
                const float qkv = acc[mi][ni][j];
                csum += (qkv + SMOOTH) / (qrow[mi * 8 + j] + kkv - qkv + SMOOTH);
            }
        atomicAdd(&colsum[colLocal], csum);
    }
    __syncthreads();
    atomicAdd(&tqk[b * NP + tileN + tid], colsum[tid]);   // 256 cols, 256 threads
}

// ---------------------------------------------------------------------------
// Stage 3: out = (tqk / 4096) * v, float4 per thread (patch width 8, 4-aligned
// chunks never straddle a patch boundary in x).
// ---------------------------------------------------------------------------
__global__ __launch_bounds__(256) void scale_kernel(const float* __restrict__ v,
                                                    const float* __restrict__ tqk,
                                                    float* __restrict__ out) {
    const size_t idx4 = (size_t)blockIdx.x * 256 + threadIdx.x;   // float4 index
    const size_t base = idx4 * 4;
    const int x  = (int)(base & 127);
    const int y  = (int)((base >> 7) & 127);
    const size_t bt = base >> 18;                                  // b*16 + t
    const float w = tqk[(bt << 8) + ((y >> 3) << 4) + (x >> 3)] * (1.0f / 4096.0f);
    const float4 vv = ((const float4*)v)[idx4];
    const float4 r = make_float4(vv.x * w, vv.y * w, vv.z * w, vv.w * w);
    ((float4*)out)[idx4] = r;
}

// ---------------------------------------------------------------------------
extern "C" void kernel_launch(void* const* d_in, const int* in_sizes, int n_in,
                              void* d_out, int out_size, void* d_ws, size_t ws_size,
                              hipStream_t stream) {
    const float* q = (const float*)d_in[0];
    const float* k = (const float*)d_in[1];
    const float* v = (const float*)d_in[2];
    float* out = (float*)d_out;

    char* ws = (char*)d_ws;
    _Float16* Ap = (_Float16*)ws;                                  // 33.55 MB
    _Float16* Bp = (_Float16*)(ws + (size_t)NB * NP * KD * 2);     // 33.55 MB
    float* qq  = (float*)(ws + (size_t)2 * NB * NP * KD * 2);
    float* kkn = qq + NB * NP;
    float* tqk = kkn + NB * NP;

    zero_kernel<<<(NB * NP + 255) / 256, 256, 0, stream>>>(tqk, NB * NP);
    pack_kernel<<<NB * NP, 256, 0, stream>>>(q, Ap, qq);
    pack_kernel<<<NB * NP, 256, 0, stream>>>(k, Bp, kkn);

    dim3 grid(NP / BN, NP / BM, NB);   // 16 x 32 x 4
    gemm_sim_kernel<<<grid, 256, 0, stream>>>(Ap, Bp, qq, kkn, tqk);

    // 4*16*16*128*128 floats / 4 per thread / 256 threads = 16384 blocks
    scale_kernel<<<16384, 256, 0, stream>>>(v, tqk, out);
}
